// CrossScaleConvolutionalAttentionDenoising_4028679324174
// MI455X (gfx1250) — matrix operand order repaired and checked by reading, not compiled, not run on hardware
//
#include <hip/hip_runtime.h>
#include <hip/hip_bf16.h>

// ---- dims ----
#define BATCH 16
#define CH    256
#define HH    64
#define WW    64
#define HW    (HH*WW)          // 4096
#define NPIX  (BATCH*HW)       // 65536
#define K3C   (3*CH)           // 768
#define EPS   1e-5f

typedef __attribute__((ext_vector_type(16))) _Float16 v16h;
typedef __attribute__((ext_vector_type(8)))  _Float16 v8h;
typedef __attribute__((ext_vector_type(8)))  float    v8f;

// ---------------------------------------------------------------------------
// Kernel 0: transpose + convert aw (ci-major f32, [768][256]) -> awT f16 [256][768]
// ---------------------------------------------------------------------------
__global__ __launch_bounds__(256)
void wprep_kernel(const float* __restrict__ aw, _Float16* __restrict__ awT) {
    int t = blockIdx.x * 256 + threadIdx.x;     // 0 .. 196607
    int co = t / K3C;
    int ci = t - co * K3C;
    awT[t] = (_Float16)aw[(size_t)ci * CH + co];
}

// ---------------------------------------------------------------------------
// Kernel 1: 3x depthwise conv + BN + SiLU.
//   writes fusedA f16 [pixel][3C]  (GEMM A operand, K contiguous)
//   writes sumS  f32 [b][c][hw]    (s0+s1+s2)
// one block = one (b,c) plane x one 16-row tile; x tile + 3-halo in LDS
// ---------------------------------------------------------------------------
__global__ __launch_bounds__(256)
void dwconv_kernel(const float* __restrict__ x,
                   const float* __restrict__ w3, const float* __restrict__ w5,
                   const float* __restrict__ w7,
                   const float* __restrict__ dwb, const float* __restrict__ bng,
                   const float* __restrict__ bnb, const float* __restrict__ bnm,
                   const float* __restrict__ bnv,
                   _Float16* __restrict__ fusedA, float* __restrict__ sumS) {
    __shared__ float tile[22 * 70];   // 16+6 rows, 64+6 cols
    __shared__ float wsm[83];         // 9 + 25 + 49 weights for this c
    const int bc  = blockIdx.x;       // b*256 + c
    const int c   = bc & (CH - 1);
    const int b   = bc >> 8;
    const int tr  = blockIdx.y;       // row tile 0..3
    const int tid = threadIdx.x;

    const float* xp = x + (size_t)bc * HW;
    const int row0 = tr * 16 - 3;
    for (int i = tid; i < 22 * 70; i += 256) {
        int r = i / 70, cc = i - r * 70;
        int gr = row0 + r, gc = cc - 3;
        float v = 0.f;
        if (gr >= 0 && gr < HH && gc >= 0 && gc < WW) v = xp[gr * WW + gc];
        tile[i] = v;
    }
    if (tid < 9)       wsm[tid] = w3[tid * CH + c];
    else if (tid < 34) wsm[tid] = w5[(tid - 9) * CH + c];
    else if (tid < 83) wsm[tid] = w7[(tid - 34) * CH + c];
    __syncthreads();

    // fold bias + BN into affine per branch: y = conv*A + B
    float A0, B0, A1, B1, A2, B2;
    {
        float s;
        s = bng[c]       * rsqrtf(bnv[c]       + EPS); A0 = s; B0 = (dwb[c]       - bnm[c])       * s + bnb[c];
        s = bng[CH + c]  * rsqrtf(bnv[CH + c]  + EPS); A1 = s; B1 = (dwb[CH + c]  - bnm[CH + c])  * s + bnb[CH + c];
        s = bng[2*CH + c]* rsqrtf(bnv[2*CH + c]+ EPS); A2 = s; B2 = (dwb[2*CH + c]- bnm[2*CH + c])* s + bnb[2*CH + c];
    }

    for (int p = tid; p < 16 * WW; p += 256) {
        int lr = p >> 6, lc = p & 63;
        float c3 = 0.f, c5 = 0.f, c7 = 0.f;
        #pragma unroll
        for (int dh = 0; dh < 3; ++dh)
            #pragma unroll
            for (int dw = 0; dw < 3; ++dw)
                c3 += tile[(lr + 2 + dh) * 70 + (lc + 2 + dw)] * wsm[dh * 3 + dw];
        #pragma unroll
        for (int dh = 0; dh < 5; ++dh)
            #pragma unroll
            for (int dw = 0; dw < 5; ++dw)
                c5 += tile[(lr + 1 + dh) * 70 + (lc + 1 + dw)] * wsm[9 + dh * 5 + dw];
        #pragma unroll
        for (int dh = 0; dh < 7; ++dh)
            #pragma unroll
            for (int dw = 0; dw < 7; ++dw)
                c7 += tile[(lr + dh) * 70 + (lc + dw)] * wsm[34 + dh * 7 + dw];

        float t0 = c3 * A0 + B0, s0 = t0 / (1.f + __expf(-t0));
        float t1 = c5 * A1 + B1, s1 = t1 / (1.f + __expf(-t1));
        float t2 = c7 * A2 + B2, s2 = t2 / (1.f + __expf(-t2));

        int h = tr * 16 + lr;
        size_t pix = (size_t)b * HW + h * WW + lc;
        _Float16* fp = fusedA + pix * K3C + c;
        fp[0]      = (_Float16)s0;
        fp[CH]     = (_Float16)s1;
        fp[2 * CH] = (_Float16)s2;
        sumS[(size_t)bc * HW + h * WW + lc] = s0 + s1 + s2;
    }
}

// ---------------------------------------------------------------------------
// Kernel 2: per-pixel mean / max over the 768 fused channels
// ---------------------------------------------------------------------------
__global__ __launch_bounds__(256)
void rowstats_kernel(const _Float16* __restrict__ fusedA,
                     float* __restrict__ avg, float* __restrict__ mx) {
    int pix = blockIdx.x * 256 + threadIdx.x;   // 0..65535
    const v8h* p = (const v8h*)(fusedA + (size_t)pix * K3C);
    float s = 0.f, m = -1e30f;
    for (int j = 0; j < K3C / 8; ++j) {
        v8h v = p[j];
        #pragma unroll
        for (int i = 0; i < 8; ++i) {
            float f = (float)v[i];
            s += f;
            m = fmaxf(m, f);
        }
    }
    avg[pix] = s * (1.0f / (float)K3C);
    mx[pix]  = m;
}

// ---------------------------------------------------------------------------
// Kernel 3: spatial attention: 3x3 conv over (avg,max) + sigmoid -> attn2[b][hw]
// ---------------------------------------------------------------------------
__global__ __launch_bounds__(256)
void spatial_kernel(const float* __restrict__ avg, const float* __restrict__ mx,
                    const float* __restrict__ sqw, const float* __restrict__ sqb,
                    float* __restrict__ attn2) {
    int pix = blockIdx.x * 256 + threadIdx.x;
    int b = pix >> 12, hw = pix & (HW - 1);
    int h = hw >> 6, w = hw & 63;
    float s = sqb[0];
    #pragma unroll
    for (int kh = 0; kh < 3; ++kh)
        #pragma unroll
        for (int kw = 0; kw < 3; ++kw) {
            int hh = h + kh - 1, ww = w + kw - 1;
            if (hh < 0 || hh >= HH || ww < 0 || ww >= WW) continue;
            int idx = (b << 12) + hh * WW + ww;
            s += avg[idx] * sqw[(kh * 3 + kw) * 2 + 0]
               + mx[idx]  * sqw[(kh * 3 + kw) * 2 + 1];
        }
    attn2[pix] = 1.f / (1.f + __expf(-s));
}

// ---------------------------------------------------------------------------
// Kernel 4: WMMA GEMM (M=65536, K=768, N=256) + fused attention epilogue.
// Block: 256 thr = 8 waves; block tile 64M x 64N; wave = 16M x 32N (2 accums).
// A/B fragments use contiguous k order k = 16*half + s (wave32); D per ISA.
// ---------------------------------------------------------------------------
__device__ __forceinline__ void epilogue_store(
    v8f acc, int c, int mStart,
    const float* __restrict__ ab,  const float* __restrict__ ag,
    const float* __restrict__ abn, const float* __restrict__ am,
    const float* __restrict__ av,
    const float* __restrict__ sumS, const float* __restrict__ attn2,
    const float* __restrict__ x, float* __restrict__ out) {
    float sA   = ag[c] * rsqrtf(av[c] + EPS);
    float sB   = abn[c] - am[c] * sA;
    float bias = ab[c];
    int b   = mStart >> 12;
    int hw0 = mStart & (HW - 1);
    size_t base = (((size_t)b * CH + c) << 12) + hw0;
    const float* sS = sumS  + base;
    const float* xp = x     + base;
    const float* a2 = attn2 + ((size_t)b << 12) + hw0;
    float* op = out + base;
    #pragma unroll
    for (int r = 0; r < 8; ++r) {
        float t = (acc[r] + bias) * sA + sB;     // 1x1 conv + bias + BN
        float u = t / (1.f + __expf(-t));        // SiLU
        float w = 1.f / (1.f + __expf(-u));      // sigmoid -> channel attention
        op[r] = w * sS[r] + a2[r] * xp[r];       // weighted + spatial*x
    }
}

__global__ __launch_bounds__(256)
void gemm_attn_kernel(const _Float16* __restrict__ fusedA,  // [65536][768]
                      const _Float16* __restrict__ awT,     // [256][768]
                      const float* __restrict__ ab,  const float* __restrict__ ag,
                      const float* __restrict__ abn, const float* __restrict__ am,
                      const float* __restrict__ av,
                      const float* __restrict__ sumS,   // [b][c][hw]
                      const float* __restrict__ attn2,  // [b][hw]
                      const float* __restrict__ x,      // [b][c][hw]
                      float* __restrict__ out) {        // [b][c][hw]
    const int lane   = threadIdx.x & 31;
    const int wave   = threadIdx.x >> 5;
    const int l15    = lane & 15;
    const int hiHalf = lane >> 4;                     // 0|1
    const int mBase  = blockIdx.x * 64 + (wave & 3) * 16;
    const int nBase  = blockIdx.y * 64 + (wave >> 2) * 32;

    // A: lane<16 holds row M=l15, K 0..15; lane>=16 holds K 16..31
    const _Float16* aRow  = fusedA + (size_t)(mBase + l15) * K3C + hiHalf * 16;
    // B: lanes 0-15 hold K 0..15 of column n; lanes 16-31 hold K 16..31
    const _Float16* bRow0 = awT + (size_t)(nBase + l15)      * K3C + hiHalf * 16;
    const _Float16* bRow1 = awT + (size_t)(nBase + 16 + l15) * K3C + hiHalf * 16;

    v8f acc0 = {}; v8f acc1 = {};
    for (int k0 = 0; k0 < K3C; k0 += 32) {
        union { v16h v; v8h h[2]; } a;
        a.h[0] = *(const v8h*)(aRow + k0);        // K[k0+koff .. +7]
        a.h[1] = *(const v8h*)(aRow + k0 + 8);    // K[k0+8+koff .. +7]
        v16h b0 = *(const v16h*)(bRow0 + k0);
        v16h b1 = *(const v16h*)(bRow1 + k0);
        acc0 = __builtin_amdgcn_wmma_f32_16x16x32_f16(false, a.v, false, b0,
                                                      (short)0, acc0, false, false);
        acc1 = __builtin_amdgcn_wmma_f32_16x16x32_f16(false, a.v, false, b1,
                                                      (short)0, acc1, false, false);
    }

    // D layout: lane<16 -> (M = mBase + r,      N = nTile + l15)
    //           lane>=16-> (M = mBase + 8 + r,  N = nTile + l15)
    const int mStart = mBase + hiHalf * 8;
    epilogue_store(acc0, nBase + l15,      mStart, ab, ag, abn, am, av, sumS, attn2, x, out);
    epilogue_store(acc1, nBase + 16 + l15, mStart, ab, ag, abn, am, av, sumS, attn2, x, out);
}

// ---------------------------------------------------------------------------
// launch
// ---------------------------------------------------------------------------
extern "C" void kernel_launch(void* const* d_in, const int* in_sizes, int n_in,
                              void* d_out, int out_size, void* d_ws, size_t ws_size,
                              hipStream_t stream) {
    // setup_inputs order, tuple dw_w flattened to 3 leaves:
    const float* x    = (const float*)d_in[0];
    const float* w3   = (const float*)d_in[1];
    const float* w5   = (const float*)d_in[2];
    const float* w7   = (const float*)d_in[3];
    const float* dwb  = (const float*)d_in[4];
    const float* bng  = (const float*)d_in[5];
    const float* bnb  = (const float*)d_in[6];
    const float* bnm  = (const float*)d_in[7];
    const float* bnv  = (const float*)d_in[8];
    const float* aw   = (const float*)d_in[9];
    const float* ab   = (const float*)d_in[10];
    const float* ag   = (const float*)d_in[11];
    const float* abn  = (const float*)d_in[12];
    const float* am   = (const float*)d_in[13];
    const float* av   = (const float*)d_in[14];
    const float* sqw  = (const float*)d_in[15];
    const float* sqb  = (const float*)d_in[16];
    float* out = (float*)d_out;

    char* ws = (char*)d_ws;
    _Float16* fusedA = (_Float16*)(ws);                                  // 100,663,296 B
    float*    sumS   = (float*)   (ws + 100663296);                      //  67,108,864 B
    _Float16* awT    = (_Float16*)(ws + 167772160);                      //     393,216 B
    float*    aggAvg = (float*)   (ws + 168165376);                      //     262,144 B
    float*    aggMax = (float*)   (ws + 168427520);                      //     262,144 B
    float*    attn2  = (float*)   (ws + 168689664);                      //     262,144 B

    wprep_kernel<<<K3C, 256, 0, stream>>>(aw, awT);
    dwconv_kernel<<<dim3(BATCH * CH, HH / 16), 256, 0, stream>>>(
        x, w3, w5, w7, dwb, bng, bnb, bnm, bnv, fusedA, sumS);
    rowstats_kernel<<<NPIX / 256, 256, 0, stream>>>(fusedA, aggAvg, aggMax);
    spatial_kernel<<<NPIX / 256, 256, 0, stream>>>(aggAvg, aggMax, sqw, sqb, attn2);
    gemm_attn_kernel<<<dim3(NPIX / 64, CH / 64), 256, 0, stream>>>(
        fusedA, awT, ab, ag, abn, am, av, sumS, attn2, x, out);
}
